// FusedExtremeClassifier_48223892799775
// MI455X (gfx1250) — compile-verified, matrix operand order checked by reading, not run on hardware
//
#include <hip/hip_runtime.h>
#include <hip/hip_bf16.h>

typedef __bf16 bf16_t;
typedef __attribute__((ext_vector_type(16))) __bf16 v16bf;
typedef __attribute__((ext_vector_type(8)))  __bf16 v8bf;
typedef __attribute__((ext_vector_type(8)))  float  v8f;
typedef __attribute__((ext_vector_type(4)))  unsigned int u32x4;
typedef __attribute__((ext_vector_type(4)))  int i32x4;
typedef __attribute__((address_space(1))) i32x4 gbl_i32x4; // global-AS int4
typedef __attribute__((address_space(3))) i32x4 lds_i32x4; // LDS-AS int4

// Problem constants (from reference)
constexpr int D    = 1024;    // in_features
constexpr int V    = 32000;   // out_features (vocab)
constexpr int NTOK = 8192;    // tokens
constexpr float SMOOTH = 0.1f;

// Tiling
constexpr int MT  = 128;          // token tile
constexpr int NT  = 128;          // vocab tile
constexpr int KT  = 32;           // k step (one WMMA K)
constexpr int NK  = D / KT;       // 32 k steps
constexpr int NMB = NTOK / MT;    // 64 token tiles
constexpr int NVB = V / NT;       // 250 vocab tiles
constexpr int SP  = 40;           // padded LDS row stride (bf16 elems) for staging
constexpr int LP  = 129;          // padded LDS row stride (floats) for logits
constexpr int STAGE_ELEMS = 4 * MT * SP;               // bf16 elems per stage buffer
constexpr int SMEM_BYTES  = 2 * STAGE_ELEMS * 2;       // double buffered: 81920 B

#if __has_builtin(__builtin_amdgcn_global_load_async_to_lds_b128) && \
    __has_builtin(__builtin_amdgcn_s_wait_asynccnt)
#define USE_ASYNC_LDS 1
#else
#define USE_ASYNC_LDS 0
#endif

static __device__ __forceinline__ v8f wmma_bf16(v16bf a, v16bf b, v8f c) {
    // 8 args: (neg_a, A, neg_b, B, c_mod, C, reuse_a, reuse_b)
    return __builtin_amdgcn_wmma_f32_16x16x32_bf16(false, a, false, b, (short)0, c, false, false);
}

#if USE_ASYNC_LDS
static __device__ __forceinline__ void async_copy16(const bf16_t* g, bf16_t* l) {
    __builtin_amdgcn_global_load_async_to_lds_b128(
        (gbl_i32x4*)g,
        (lds_i32x4*)l,
        0, 0);
}
#endif

// ---------------- Pass 1: f32 -> bf16 hi/lo split ----------------
__global__ void convert_split_kernel(const float* __restrict__ src,
                                     bf16_t* __restrict__ hi,
                                     bf16_t* __restrict__ lo,
                                     size_t n) {
    size_t stride = (size_t)gridDim.x * blockDim.x;
    for (size_t i = (size_t)blockIdx.x * blockDim.x + threadIdx.x; i < n; i += stride) {
        float v = src[i];
        bf16_t h = (bf16_t)v;
        hi[i] = h;
        lo[i] = (bf16_t)(v - (float)h);
    }
}

// ---------------- Pass 2: fused GEMM + bias + per-tile softmax partials ----------------
__global__ __launch_bounds__(256, 1)
void gemm_partials_kernel(const bf16_t* __restrict__ xh, const bf16_t* __restrict__ xl,
                          const bf16_t* __restrict__ wh, const bf16_t* __restrict__ wl,
                          const float*  __restrict__ bias,
                          const int*    __restrict__ target,
                          float* __restrict__ partM, float* __restrict__ partS,
                          float* __restrict__ partZ, float* __restrict__ tgtLogit) {
    __shared__ __align__(16) unsigned char smem[SMEM_BYTES]; // 2 stage buffers; buf0 reused for logits

    const int tid  = threadIdx.x;
    const int lane = tid & 31;
    const int wid  = tid >> 5;          // 8 waves
    const int wm   = wid >> 2;          // 0..1  : 64-row half
    const int wn   = wid & 3;           // 0..3  : 32-col slice
    const int mBase = blockIdx.x * MT;  // token base
    const int vBase = blockIdx.y * NT;  // vocab base

    v8f c[4][2];
    {
        v8f z = {};
        for (int i = 0; i < 4; ++i)
            for (int j = 0; j < 2; ++j) c[i][j] = z;
    }

    // Stage 128 rows x 32 cols bf16 of all four operand arrays into LDS buffer `buf`.
    // 512 16B chunks per array, 256 threads -> 2 chunks per array per thread.
    auto stage = [&](int kBase, int buf) {
        bf16_t* base = (bf16_t*)smem + buf * STAGE_ELEMS;
        bf16_t* dXh = base;
        bf16_t* dXl = dXh + MT * SP;
        bf16_t* dWh = dXl + MT * SP;
        bf16_t* dWl = dWh + NT * SP;
        for (int ch = tid; ch < 512; ch += 256) {
            int row = ch >> 2;
            int kc  = (ch & 3) << 3; // 0,8,16,24
            size_t gx = (size_t)(mBase + row) * D + kBase + kc;
            size_t gw = (size_t)(vBase + row) * D + kBase + kc;
            int loff = row * SP + kc;
#if USE_ASYNC_LDS
            async_copy16(xh + gx, dXh + loff);
            async_copy16(xl + gx, dXl + loff);
            async_copy16(wh + gw, dWh + loff);
            async_copy16(wl + gw, dWl + loff);
#else
            *(u32x4*)(dXh + loff) = *(const u32x4*)(xh + gx);
            *(u32x4*)(dXl + loff) = *(const u32x4*)(xl + gx);
            *(u32x4*)(dWh + loff) = *(const u32x4*)(wh + gw);
            *(u32x4*)(dWl + loff) = *(const u32x4*)(wl + gw);
#endif
        }
    };

    // fragment load: two contiguous 16B chunks per lane at K = hi*8 and 16+hi*8
    auto loadFrag = [&](const bf16_t* s, int rowBase) -> v16bf {
        int r  = rowBase + (lane & 15);
        int hi = lane >> 4;
        v8bf a0 = *(const v8bf*)(s + r * SP + hi * 8);
        v8bf a1 = *(const v8bf*)(s + r * SP + 16 + hi * 8);
        return __builtin_shufflevector(a0, a1, 0, 1, 2, 3, 4, 5, 6, 7, 8, 9, 10, 11, 12, 13, 14, 15);
    };

    stage(0, 0); // prologue fill of buffer 0

    for (int ks = 0; ks < NK; ++ks) {
        const int cur = ks & 1;
#if USE_ASYNC_LDS
        __builtin_amdgcn_s_wait_asynccnt(0); // own async stores to LDS done
#endif
        __syncthreads();                     // everyone's fill of buf[cur] visible; reads of buf[cur^1] done
        if (ks + 1 < NK) stage((ks + 1) * KT, cur ^ 1); // overlap next fill with compute

        bf16_t* base = (bf16_t*)smem + cur * STAGE_ELEMS;
        const bf16_t* sXh = base;
        const bf16_t* sXl = sXh + MT * SP;
        const bf16_t* sWh = sXl + MT * SP;
        const bf16_t* sWl = sWh + NT * SP;

        // preload all fragments for max ILP across the 24 WMMAs
        v16bf bh[2], bl[2], ah[4], al[4];
        for (int j = 0; j < 2; ++j) {
            bh[j] = loadFrag(sWh, wn * 32 + j * 16);
            bl[j] = loadFrag(sWl, wn * 32 + j * 16);
        }
        for (int i = 0; i < 4; ++i) {
            ah[i] = loadFrag(sXh, wm * 64 + i * 16);
            al[i] = loadFrag(sXl, wm * 64 + i * 16);
        }
        for (int i = 0; i < 4; ++i)
            for (int j = 0; j < 2; ++j) {
                c[i][j] = wmma_bf16(ah[i], bh[j], c[i][j]); // hi*hi
                c[i][j] = wmma_bf16(ah[i], bl[j], c[i][j]); // hi*lo
                c[i][j] = wmma_bf16(al[i], bh[j], c[i][j]); // lo*hi
            }
    }
    __syncthreads(); // staging dead; reuse LDS (buf0 region) for logits

    float b0[2];
    for (int j = 0; j < 2; ++j)
        b0[j] = bias[vBase + wn * 32 + j * 16 + (lane & 15)];

    float* sLog = (float*)smem; // 64 x LP floats (33024 B < stage buffer 0)
    for (int phase = 0; phase < 2; ++phase) {
        if (wm == phase) {
            // C/D layout: VGPR r -> M = r + 8*(lane>=16); N = lane%16
            for (int i = 0; i < 4; ++i)
                for (int j = 0; j < 2; ++j) {
                    int colL = wn * 32 + j * 16 + (lane & 15);
                    int rB   = i * 16 + 8 * (lane >> 4);
                    for (int r = 0; r < 8; ++r)
                        sLog[(rB + r) * LP + colL] = c[i][j][r] + b0[j];
                }
        }
        __syncthreads();
        if (tid < 64) {
            int row   = tid;
            int token = mBase + phase * 64 + row;
            const float* lr = sLog + row * LP;
            float m = -__builtin_inff();
            for (int cc = 0; cc < NT; ++cc) m = fmaxf(m, lr[cc]);
            float s = 0.0f, z = 0.0f;
            for (int cc = 0; cc < NT; ++cc) {
                float v = lr[cc];
                s += __expf(v - m);
                z += v;
            }
            size_t pidx = (size_t)token * NVB + blockIdx.y;
            partM[pidx] = m;
            partS[pidx] = s;
            partZ[pidx] = z;
            int tg = target[token];
            if (tg >= vBase && tg < vBase + NT)
                tgtLogit[token] = lr[tg - vBase]; // exactly one tile owns it
        }
        __syncthreads();
    }
}

// ---------------- Pass 3: per-token logsumexp + smoothed NLL ----------------
__global__ __launch_bounds__(256, 1)
void token_reduce_kernel(const float* __restrict__ partM, const float* __restrict__ partS,
                         const float* __restrict__ partZ, const float* __restrict__ tgtLogit,
                         const int* __restrict__ target, float* __restrict__ perTok) {
    __shared__ float red[256];
    int token = blockIdx.x;
    int tid   = threadIdx.x;
    size_t base = (size_t)token * NVB;

    float m = (tid < NVB) ? partM[base + tid] : -__builtin_inff();
    red[tid] = m;
    __syncthreads();
    for (int s = 128; s; s >>= 1) {
        if (tid < s) red[tid] = fmaxf(red[tid], red[tid + s]);
        __syncthreads();
    }
    float M = red[0];
    __syncthreads();

    float sacc = 0.0f, zacc = 0.0f;
    if (tid < NVB) {
        sacc = partS[base + tid] * __expf(m - M);
        zacc = partZ[base + tid];
    }
    red[tid] = sacc;
    __syncthreads();
    for (int s = 128; s; s >>= 1) {
        if (tid < s) red[tid] += red[tid + s];
        __syncthreads();
    }
    float S = red[0];
    __syncthreads();
    red[tid] = zacc;
    __syncthreads();
    for (int s = 128; s; s >>= 1) {
        if (tid < s) red[tid] += red[tid + s];
        __syncthreads();
    }
    if (tid == 0) {
        float Z   = red[0];
        float lse = M + __logf(S);
        float zt  = tgtLogit[token];
        float pt  = lse - (1.0f - SMOOTH) * zt - (SMOOTH / (float)V) * Z;
        perTok[token] = (target[token] != 0) ? pt : 0.0f;
    }
}

// ---------------- Pass 4: deterministic final sum ----------------
__global__ __launch_bounds__(256, 1)
void final_sum_kernel(const float* __restrict__ perTok, float* __restrict__ out) {
    __shared__ float red[256];
    int tid = threadIdx.x;
    float a = 0.0f;
    for (int t = tid; t < NTOK; t += 256) a += perTok[t]; // fixed order
    red[tid] = a;
    __syncthreads();
    for (int s = 128; s; s >>= 1) {
        if (tid < s) red[tid] += red[tid + s];
        __syncthreads();
    }
    if (tid == 0) out[0] = red[0];
}

extern "C" void kernel_launch(void* const* d_in, const int* in_sizes, int n_in,
                              void* d_out, int out_size, void* d_ws, size_t ws_size,
                              hipStream_t stream) {
    const float* x      = (const float*)d_in[0]; // [NTOK, D]
    const int*   target = (const int*)d_in[1];   // [NTOK]
    const float* w      = (const float*)d_in[2]; // [V, D]
    const float* bias   = (const float*)d_in[3]; // [V]
    float* out = (float*)d_out;

    // Workspace carve-out (256B aligned)
    char* ws = (char*)d_ws;
    size_t off = 0;
    auto carve = [&](size_t bytes) -> char* {
        char* p = ws + off;
        off = (off + bytes + 255) & ~(size_t)255;
        return p;
    };
    bf16_t* xh = (bf16_t*)carve((size_t)NTOK * D * sizeof(bf16_t));
    bf16_t* xl = (bf16_t*)carve((size_t)NTOK * D * sizeof(bf16_t));
    bf16_t* wh = (bf16_t*)carve((size_t)V * D * sizeof(bf16_t));
    bf16_t* wl = (bf16_t*)carve((size_t)V * D * sizeof(bf16_t));
    float* partM    = (float*)carve((size_t)NTOK * NVB * sizeof(float));
    float* partS    = (float*)carve((size_t)NTOK * NVB * sizeof(float));
    float* partZ    = (float*)carve((size_t)NTOK * NVB * sizeof(float));
    float* tgtLogit = (float*)carve((size_t)NTOK * sizeof(float));
    float* perTok   = (float*)carve((size_t)NTOK * sizeof(float));

    // Pass 1: split conversions
    convert_split_kernel<<<2048, 256, 0, stream>>>(x, xh, xl, (size_t)NTOK * D);
    convert_split_kernel<<<4096, 256, 0, stream>>>(w, wh, wl, (size_t)V * D);

    // Pass 2: fused GEMM + partial softmax stats
    dim3 grid(NMB, NVB); // 64 x 250
    gemm_partials_kernel<<<grid, 256, 0, stream>>>(xh, xl, wh, wl, bias, target,
                                                   partM, partS, partZ, tgtLogit);

    // Pass 3: per-token reduction
    token_reduce_kernel<<<NTOK, 256, 0, stream>>>(partM, partS, partZ, tgtLogit, target, perTok);

    // Pass 4: deterministic scalar sum
    final_sum_kernel<<<1, 256, 0, stream>>>(perTok, out);
}
